// GaussianImage_Cholesky_EA_21663815041286
// MI455X (gfx1250) — matrix operand order repaired and checked by reading, not compiled, not run on hardware
//
#include <hip/hip_runtime.h>

typedef __attribute__((ext_vector_type(16))) _Float16 v16h;
typedef __attribute__((ext_vector_type(8)))  float    v8f;

#define NG    2048
#define RANKF 12
#define IMG_W 128
#define IMG_H 128

// 32-byte per-gaussian parameter block: two b128 accesses.
struct __align__(16) GParam {
  float cx, cy, ca, cb;        // first b128
  float cc, op;                // second b128 ...
  unsigned int mlo, mhi;       // 64-bit 8x8 tile-inside mask
};

// ---------------------------------------------------------------------------
// Phase 1: per-gaussian parameters (conic, center, opacity, tile mask)
// ---------------------------------------------------------------------------
__global__ void prep_gaussians(const float* __restrict__ xyz,
                               const float* __restrict__ chol,
                               const float* __restrict__ opacity,
                               GParam* __restrict__ params) {
  int g = blockIdx.x * blockDim.x + threadIdx.x;
  if (g >= NG) return;
  float mx = tanhf(xyz[g * 2 + 0]);
  float my = tanhf(xyz[g * 2 + 1]);
  float l1 = chol[g * 3 + 0] + 0.5f;
  float l2 = chol[g * 3 + 1];
  float l3 = chol[g * 3 + 2] + 0.5f;
  float c00 = l1 * l1;
  float c01 = l1 * l2;
  float c11 = l2 * l2 + l3 * l3;
  float det = c00 * c11 - c01 * c01;
  float inv = 1.0f / det;
  float cxv = 0.5f * (float)IMG_W * (mx + 1.0f);
  float cyv = 0.5f * (float)IMG_H * (my + 1.0f);
  float mid = 0.5f * (c00 + c11);
  float v1  = mid + sqrtf(fmaxf(mid * mid - det, 0.1f));
  float radius = ceilf(3.0f * sqrtf(v1));
  int tminx = (int)fminf(fmaxf(floorf((cxv - radius) * (1.0f / 16.0f)), 0.0f), 8.0f);
  int tmaxx = (int)fminf(fmaxf(floorf((cxv + radius) * (1.0f / 16.0f)) + 1.0f, 0.0f), 8.0f);
  int tminy = (int)fminf(fmaxf(floorf((cyv - radius) * (1.0f / 16.0f)), 0.0f), 8.0f);
  int tmaxy = (int)fminf(fmaxf(floorf((cyv + radius) * (1.0f / 16.0f)) + 1.0f, 0.0f), 8.0f);

  // 8x8 tile mask: bit (ty*8+tx) set iff tile is inside [tmin,tmax) ranges.
  unsigned int rowmask = (1u << tmaxx) - (1u << tminx);  // empty range -> 0
  unsigned long long m = 0ull;
#pragma unroll
  for (int ty = 0; ty < 8; ++ty)
    if (ty >= tminy && ty < tmaxy)
      m |= ((unsigned long long)rowmask) << (8 * ty);

  GParam gp;
  gp.cx = cxv; gp.cy = cyv;
  gp.ca = c11 * inv; gp.cb = -c01 * inv;
  gp.cc = c00 * inv; gp.op = opacity[g];
  gp.mlo = (unsigned int)(m & 0xffffffffull);
  gp.mhi = (unsigned int)(m >> 32);
  params[g] = gp;
}

// ---------------------------------------------------------------------------
// Phase 2: pack features into per-lane-contiguous B panels.
// bpack[((chunk*2 + half)*16 + n)*8 + r] = {f16(F[k][n]), f16(F[k+1][n])},
// k = chunk*32 + half*16 + 2r.  A lane's 8 dwords per chunk are a contiguous
// 32B run -> two global_load_b128; the wave reads 1KB contiguous.
// ---------------------------------------------------------------------------
__global__ void pack_features(const float* __restrict__ feat,
                              unsigned int* __restrict__ bpack) {
  int idx = blockIdx.x * blockDim.x + threadIdx.x;   // 0 .. 16383
  if (idx >= (NG / 32) * 2 * 16 * 8) return;
  int r = idx & 7;
  int n = (idx >> 3) & 15;
  int h = (idx >> 7) & 1;
  int c = idx >> 8;
  int k = c * 32 + h * 16 + 2 * r;
  float f0 = (n < RANKF) ? feat[(k + 0) * RANKF + n] : 0.0f;
  float f1 = (n < RANKF) ? feat[(k + 1) * RANKF + n] : 0.0f;
  union { _Float16 hh[2]; unsigned int u; } p;
  p.hh[0] = (_Float16)f0;
  p.hh[1] = (_Float16)f1;
  bpack[idx] = p.u;
}

// ---------------------------------------------------------------------------
// Phase 3: render. One wave32 per 16-pixel M-tile (all 16 pixels share one
// image tile -> tile test is wave-uniform and folded into staged opacity).
// Wave-private LDS staging (no barriers; same-wave DS ordering), register
// double-buffered param prefetch, reduction via v_wmma_f32_16x16x32_f16.
// ---------------------------------------------------------------------------
__global__ void __launch_bounds__(128)
render(const GParam* __restrict__ params,
       const unsigned int* __restrict__ bpack,
       float* __restrict__ out) {
  __shared__ GParam sg[4][32];                 // per-wave private buffers

  const int tid     = threadIdx.x;
  const int lane    = tid & 31;
  const int waveId  = blockIdx.x * 4 + (tid >> 5);
  const int pixBase = waveId * 16;
  GParam* __restrict__ myg = sg[tid >> 5];

  // A layout: lane holds row M = lane&15; lanes 0-15 carry K {0..15},
  // lanes 16-31 carry K {16..31} of each 32-gaussian chunk.
  const int   m       = lane & 15;
  const int   p       = pixBase + m;
  const float pxf     = (float)(p & (IMG_W - 1));
  const float pyf     = (float)(p >> 7);
  // Wave-uniform tile id (16 aligned pixels of one row = one 16x16 tile).
  const int   tileBit = (((p >> 7) >> 4) << 3) | ((p & (IMG_W - 1)) >> 4);
  const int   selHi   = (tileBit & 32) != 0;
  const int   shAmt   = tileBit & 31;
  const int   klo     = (lane < 16) ? 0 : 16;
  const int   n       = lane & 15;             // B/D column for this lane
  const int   bOff    = (((lane >> 4) * 16 + n) << 3);  // per-lane B offset

  v8f acc = {};
  GParam cur = params[lane];                   // chunk 0 prefetch

  for (int kb = 0; kb < NG; kb += 32) {
    // Fold this wave's tile-inside test into the staged opacity: a miss
    // forces alpha = 0 < 1/255 so the weight drops out for free.
    unsigned int msel = selHi ? cur.mhi : cur.mlo;
    if (((msel >> shAmt) & 1u) == 0u) cur.op = 0.0f;
    myg[lane] = cur;                           // 2x ds_store_b128 (own slot)

    // Prefetch next chunk's gaussian while this chunk computes.
    int gn = kb + 32 + lane;
    cur = params[(gn < NG) ? gn : lane];

    // A fragment: 16 weights (this lane's pixel x 16 of the chunk's 32 K's).
    v16h a;
#pragma unroll
    for (int j = 0; j < 16; ++j) {
      const int kl = klo + j;
      GParam gp = myg[kl];                     // broadcast ds_load
      float dx = gp.cx - pxf;
      float dy = gp.cy - pyf;
      float sigma = 0.5f * (gp.ca * dx * dx + gp.cc * dy * dy)
                  + gp.cb * dx * dy;
      // sigma<0 is gated below, so the exp clamp is unnecessary; min_num
      // absorbs the op=0 * inf corner to 0.999 which the gate rejects.
      float alpha = fminf(0.999f, gp.op * __expf(-sigma));
      int valid = (int)(sigma >= 0.0f) & (int)(alpha >= (1.0f / 255.0f));
      a[j] = (_Float16)(valid ? alpha : 0.0f);
    }

    // B fragment: two coalesced b128 loads from the per-lane-contiguous panel.
    const unsigned int* bp = bpack + (kb << 3) + bOff;
    union { unsigned int u[8]; v16h h; } bb;
#pragma unroll
    for (int r = 0; r < 8; ++r)
      bb.u[r] = bp[r];

    acc = __builtin_amdgcn_wmma_f32_16x16x32_f16(
        /*neg_a=*/false, a, /*neg_b=*/false, bb.h,
        /*c_mod=*/(short)0, acc, /*reuse_a=*/false, /*reuse_b=*/false);
  }

  // D layout: lane holds column N = lane&15; VGPR i -> row M = mBase + i.
  const int mBase = (lane < 16) ? 0 : 8;
  if (n < RANKF) {
#pragma unroll
    for (int i = 0; i < 8; ++i) {
      float v = fminf(fmaxf(acc[i], 0.0f), 1.0f);
      out[(pixBase + mBase + i) * RANKF + n] = v;
    }
  }
}

// ---------------------------------------------------------------------------
extern "C" void kernel_launch(void* const* d_in, const int* in_sizes, int n_in,
                              void* d_out, int out_size, void* d_ws, size_t ws_size,
                              hipStream_t stream) {
  (void)in_sizes; (void)n_in; (void)out_size; (void)ws_size;
  const float* xyz  = (const float*)d_in[0];   // (N,2)
  const float* chol = (const float*)d_in[1];   // (N,3)
  const float* feat = (const float*)d_in[2];   // (N,12)
  const float* opa  = (const float*)d_in[3];   // (N,1)

  char* ws = (char*)d_ws;
  GParam*       params = (GParam*)ws;                    // 2048 * 32B = 64KB
  unsigned int* bpack  = (unsigned int*)(ws + 65536);    // 16384 dwords = 64KB

  prep_gaussians<<<8, 256, 0, stream>>>(xyz, chol, opa, params);
  pack_features<<<64, 256, 0, stream>>>(feat, bpack);
  render<<<256, 128, 0, stream>>>(params, bpack, (float*)d_out);
}
